// MultiheadSelfAttention_49701361549423
// MI455X (gfx1250) — matrix operand order repaired and checked by reading, not compiled, not run on hardware
//
#include <hip/hip_runtime.h>

#define BB 2
#define SS 2048
#define DD 1024
#define HH 16
#define DK 64

typedef __attribute__((ext_vector_type(16))) __bf16 v16bf;
typedef __attribute__((ext_vector_type(8)))  __bf16 v8bf;
typedef __attribute__((ext_vector_type(8)))  float  v8f;

static __device__ __forceinline__ v8f wmma_bf16(v16bf a, v16bf b, v8f c) {
    // D = A(16x32 bf16) * B(32x16 bf16) + C(16x16 f32)
    return __builtin_amdgcn_wmma_f32_16x16x32_bf16(
        /*neg_a=*/false, a, /*neg_b=*/false, b,
        /*c_mod=*/(short)0, c, /*reuse_a=*/false, /*reuse_b=*/false);
}

// Async copy 16B/lane from global to LDS (ASYNCcnt-tracked, CDNA5 path).
static __device__ __forceinline__ void async_g2l_b128(unsigned lds_off,
                                                      const void* gptr) {
    asm volatile("global_load_async_to_lds_b128 %0, %1, off"
                 :: "v"(lds_off), "v"((unsigned long long)(size_t)gptr)
                 : "memory");
}

// ---------------------------------------------------------------- fp32 -> bf16
__global__ __launch_bounds__(256) void cvt_kernel(const float* __restrict__ s,
                                                  __bf16* __restrict__ d, int n) {
    int i = blockIdx.x * blockDim.x + threadIdx.x;
    if (i < n) d[i] = (__bf16)s[i];
}

// -------------------------------------------- QKV projection + RoPE (bf16 WMMA)
// One wave computes a 16(M) x 64(N) strip == one head's full dk width.
__global__ __launch_bounds__(256) void qkv_rope_kernel(
    const __bf16* __restrict__ xb, const __bf16* __restrict__ Wq,
    const __bf16* __restrict__ Wk, const __bf16* __restrict__ Wv,
    __bf16* __restrict__ Qb, __bf16* __restrict__ Kb, __bf16* __restrict__ Vt) {
    const int lane = threadIdx.x & 31;
    const int wv   = threadIdx.x >> 5;
    const int lh = lane & 15, hig = lane >> 4;
    const int m0 = blockIdx.x * 16;
    const int h  = blockIdx.y * 8 + wv;       // head == 64-col strip
    const int n0 = h * 64;
    const int z  = blockIdx.z;
    const __bf16* W = (z == 0) ? Wq : (z == 1) ? Wk : Wv;
    const __bf16* arow = xb + (size_t)(m0 + lh) * DD;
    v8f acc[4];
    v8f zero = {};
#pragma unroll
    for (int c = 0; c < 4; ++c) acc[c] = zero;
    for (int k = 0; k < DD; k += 32) {
        v8bf a0 = *(const v8bf*)(arow + k + hig * 16);
        v8bf a1 = *(const v8bf*)(arow + k + 8 + hig * 16);
        v16bf a;
#pragma unroll
        for (int e = 0; e < 8; ++e) { a[e] = a0[e]; a[8 + e] = a1[e]; }
#pragma unroll
        for (int c = 0; c < 4; ++c) {
            const __bf16* brow = W + (size_t)(n0 + c * 16 + lh) * DD;
            v16bf b = *(const v16bf*)(brow + k + hig * 16);
            acc[c] = wmma_bf16(a, b, acc[c]);
        }
    }
    const int bi = m0 >> 11;                  // batch
    const int s0 = m0 & (SS - 1);
    if (z < 2) {
        __bf16* dst = (z == 0) ? Qb : Kb;
#pragma unroll
        for (int c = 0; c < 4; ++c) {
            const int j = c * 16 + lh;        // dim within head
            const float freq = __expf(-(float)(2 * (j >> 1)) * (9.210340371976184f / 64.f));
            const float sgn  = (j & 1) ? 1.f : -1.f;
#pragma unroll
            for (int r = 0; r < 8; ++r) {
                const int srow = s0 + r + hig * 8;
                float xv  = acc[c][r];
                float pv  = __shfl_xor(xv, 1, 32);   // RoPE pair partner
                float ang = (float)srow * freq;
                float out = xv * __cosf(ang) + sgn * pv * __sinf(ang);
                dst[(((size_t)bi * HH + h) * SS + srow) * DK + j] = (__bf16)out;
            }
        }
    } else {
#pragma unroll
        for (int c = 0; c < 4; ++c) {
            const int j = c * 16 + lh;
#pragma unroll
            for (int r = 0; r < 8; ++r) {
                const int srow = s0 + r + hig * 8;
                Vt[(((size_t)bi * HH + h) * DK + j) * SS + srow] = (__bf16)acc[c][r];
            }
        }
    }
}

// --------------------------------------------------- causal flash attention
// Block = 64 queries (4 waves x 16). Uniform trip count nkb = blockIdx.x+1:
// only the final (diagonal) 64x64 tile needs masking. K/V tiles are staged
// once per BLOCK into double-buffered LDS with global_load_async_to_lds_b128,
// overlapped with compute; fragments then come from LDS (ds_load_b128).
__global__ __launch_bounds__(128) void attn_kernel(
    const __bf16* __restrict__ Qb, const __bf16* __restrict__ Kb,
    const __bf16* __restrict__ Vt, __bf16* __restrict__ ctx) {
    __shared__ __align__(128) __bf16 ktile[2][64 * 64];   // [key][dk]
    __shared__ __align__(128) __bf16 vtile[2][64 * 64];   // [dk][key]
    __shared__ __align__(32)  __bf16 pbuf[4][16 * 64];
    const int lane = threadIdx.x & 31;
    const int wv   = threadIdx.x >> 5;
    const int lh = lane & 15, hig = lane >> 4;
    const int q0 = (blockIdx.x * 4 + wv) * 16;
    const int h = blockIdx.y, b = blockIdx.z;
    const size_t bh = (size_t)b * HH + h;
    const __bf16* Qp = Qb + bh * SS * DK;
    const __bf16* Kp = Kb + bh * SS * DK;
    const __bf16* Vp = Vt + bh * DK * SS;
    __bf16* pb = pbuf[wv];

    // issue one 64x64 K tile + one 64x64 V^T tile: 8 async b128 per wave
    auto issue_tile = [&](int kb, int buf) {
        const int k0 = kb * 64;
#pragma unroll
        for (int i = 0; i < 4; ++i) {
            const int ch  = (wv * 4 + i) * 32 + lane;  // 0..511 over block
            const int row = ch >> 3;                   // tile row (key or dk)
            const int cc  = ch & 7;                    // 16B chunk within row
            async_g2l_b128((unsigned)(size_t)&ktile[buf][row * 64 + cc * 8],
                           Kp + (size_t)(k0 + row) * DK + cc * 8);
            async_g2l_b128((unsigned)(size_t)&vtile[buf][row * 64 + cc * 8],
                           Vp + (size_t)row * SS + k0 + cc * 8);
        }
    };

    // Q A-fragments: dk chunks [0,32) and [32,64) -- resident all iterations
    v16bf aQ0, aQ1;
    {
        const __bf16* qr = Qp + (size_t)(q0 + lh) * DK;
        v8bf t0 = *(const v8bf*)(qr + hig * 16);
        v8bf t1 = *(const v8bf*)(qr + 8 + hig * 16);
        v8bf t2 = *(const v8bf*)(qr + 32 + hig * 16);
        v8bf t3 = *(const v8bf*)(qr + 40 + hig * 16);
#pragma unroll
        for (int e = 0; e < 8; ++e) {
            aQ0[e] = t0[e]; aQ0[8 + e] = t1[e];
            aQ1[e] = t2[e]; aQ1[8 + e] = t3[e];
        }
    }
    float mrow[8], lrow[8];
    v8f zero = {};
    v8f oacc[4];
#pragma unroll
    for (int r = 0; r < 8; ++r) { mrow[r] = -1e30f; lrow[r] = 0.f; }
#pragma unroll
    for (int c = 0; c < 4; ++c) oacc[c] = zero;

    const int nkb = blockIdx.x + 1;          // uniform across the block
    issue_tile(0, 0);
    for (int kb = 0; kb < nkb; ++kb) {
        const int buf = kb & 1;
        if (kb + 1 < nkb) {
            issue_tile(kb + 1, buf ^ 1);
            // async loads complete in order: <=8 outstanding => tile kb done
            asm volatile("s_wait_asynccnt 8" ::: "memory");
        } else {
            asm volatile("s_wait_asynccnt 0" ::: "memory");
        }
        __syncthreads();
        const __bf16* kt = ktile[buf];
        const __bf16* vt = vtile[buf];
        // scores: 16 q-rows x 64 keys (4 N-groups x 2 dk-chunks) from LDS
        v8f sc[4];
#pragma unroll
        for (int g = 0; g < 4; ++g) {
            const __bf16* kr = kt + (g * 16 + lh) * 64;
            v16bf bk0 = *(const v16bf*)(kr + hig * 16);
            v16bf bk1 = *(const v16bf*)(kr + 32 + hig * 16);
            v8f a = zero;
            a = wmma_bf16(aQ0, bk0, a);
            a = wmma_bf16(aQ1, bk1, a);
            sc[g] = a;
        }
        if (kb == nkb - 1) {                 // diagonal tile: causal mask
            const int k0 = kb * 64;
#pragma unroll
            for (int g = 0; g < 4; ++g) {
                const int key = k0 + g * 16 + lh;
#pragma unroll
                for (int r = 0; r < 8; ++r) {
                    const int qrow = q0 + r + hig * 8;
                    sc[g][r] = (key <= qrow) ? sc[g][r] * 0.125f : -1e30f;
                }
            }
        } else {
#pragma unroll
            for (int g = 0; g < 4; ++g)
#pragma unroll
                for (int r = 0; r < 8; ++r) sc[g][r] *= 0.125f;
        }
        // online softmax (row = 16 lanes of one half-wave)
        float alpha[8];
#pragma unroll
        for (int r = 0; r < 8; ++r) {
            float v = fmaxf(fmaxf(sc[0][r], sc[1][r]), fmaxf(sc[2][r], sc[3][r]));
            v = fmaxf(v, __shfl_xor(v, 1, 16));
            v = fmaxf(v, __shfl_xor(v, 2, 16));
            v = fmaxf(v, __shfl_xor(v, 4, 16));
            v = fmaxf(v, __shfl_xor(v, 8, 16));
            float mn = fmaxf(mrow[r], v);
            alpha[r] = __expf(mrow[r] - mn);
            mrow[r] = mn;
        }
#pragma unroll
        for (int r = 0; r < 8; ++r) {
            float t = 0.f;
#pragma unroll
            for (int g = 0; g < 4; ++g) {
                float p = __expf(sc[g][r] - mrow[r]);
                sc[g][r] = p;
                t += p;
            }
            t += __shfl_xor(t, 1, 16);
            t += __shfl_xor(t, 2, 16);
            t += __shfl_xor(t, 4, 16);
            t += __shfl_xor(t, 8, 16);
            lrow[r] = lrow[r] * alpha[r] + t;
        }
#pragma unroll
        for (int c = 0; c < 4; ++c)
#pragma unroll
            for (int r = 0; r < 8; ++r) oacc[c][r] *= alpha[r];
        // C-fragment -> A-fragment transpose of P (16x64) through per-wave LDS
#pragma unroll
        for (int g = 0; g < 4; ++g)
#pragma unroll
            for (int r = 0; r < 8; ++r)
                pb[(r + hig * 8) * 64 + g * 16 + lh] = (__bf16)sc[g][r];
        v16bf aP0, aP1;
        {
            const __bf16* pr = pb + lh * 64;
            v8bf t0 = *(const v8bf*)(pr + hig * 16);
            v8bf t1 = *(const v8bf*)(pr + 8 + hig * 16);
            v8bf t2 = *(const v8bf*)(pr + 32 + hig * 16);
            v8bf t3 = *(const v8bf*)(pr + 40 + hig * 16);
#pragma unroll
            for (int e = 0; e < 8; ++e) {
                aP0[e] = t0[e]; aP0[8 + e] = t1[e];
                aP1[e] = t2[e]; aP1[8 + e] = t3[e];
            }
        }
        // O += P V from LDS V^T tile (local key index)
#pragma unroll
        for (int c = 0; c < 4; ++c) {
            const __bf16* vr = vt + (c * 16 + lh) * 64;
            v16bf bv0 = *(const v16bf*)(vr + hig * 16);        // keys [0,32)
            v16bf bv1 = *(const v16bf*)(vr + 32 + hig * 16);   // keys [32,64)
            oacc[c] = wmma_bf16(aP0, bv0, oacc[c]);
            oacc[c] = wmma_bf16(aP1, bv1, oacc[c]);
        }
        __syncthreads();   // tile kb fully consumed before its buffer is reused
    }
    // normalize and store context [B,S,H,dk] == [N,D] bf16
#pragma unroll
    for (int c = 0; c < 4; ++c)
#pragma unroll
        for (int r = 0; r < 8; ++r) {
            const int row = q0 + r + hig * 8;
            const float val = oacc[c][r] / lrow[r];
            ctx[((size_t)(b * SS + row)) * DD + h * DK + c * 16 + lh] = (__bf16)val;
        }
}

// -------------------------------------------------- output projection (fp32 out)
__global__ __launch_bounds__(256) void oproj_kernel(
    const __bf16* __restrict__ ctx, const __bf16* __restrict__ Wo,
    float* __restrict__ out) {
    const int lane = threadIdx.x & 31;
    const int wv   = threadIdx.x >> 5;
    const int lh = lane & 15, hig = lane >> 4;
    const int m0 = blockIdx.x * 16;
    const int n0 = (blockIdx.y * 8 + wv) * 64;
    const __bf16* arow = ctx + (size_t)(m0 + lh) * DD;
    v8f acc[4];
    v8f zero = {};
#pragma unroll
    for (int c = 0; c < 4; ++c) acc[c] = zero;
    for (int k = 0; k < DD; k += 32) {
        v8bf a0 = *(const v8bf*)(arow + k + hig * 16);
        v8bf a1 = *(const v8bf*)(arow + k + 8 + hig * 16);
        v16bf a;
#pragma unroll
        for (int e = 0; e < 8; ++e) { a[e] = a0[e]; a[8 + e] = a1[e]; }
#pragma unroll
        for (int c = 0; c < 4; ++c) {
            const __bf16* brow = Wo + (size_t)(n0 + c * 16 + lh) * DD;
            v16bf b = *(const v16bf*)(brow + k + hig * 16);
            acc[c] = wmma_bf16(a, b, acc[c]);
        }
    }
#pragma unroll
    for (int c = 0; c < 4; ++c)
#pragma unroll
        for (int r = 0; r < 8; ++r)
            out[(size_t)(m0 + r + hig * 8) * DD + n0 + c * 16 + lh] = acc[c][r];
}

extern "C" void kernel_launch(void* const* d_in, const int* in_sizes, int n_in,
                              void* d_out, int out_size, void* d_ws, size_t ws_size,
                              hipStream_t stream) {
    const float* x  = (const float*)d_in[0];
    const float* Wq = (const float*)d_in[1];
    const float* Wk = (const float*)d_in[2];
    const float* Wv = (const float*)d_in[3];
    const float* Wo = (const float*)d_in[4];

    char* ws = (char*)d_ws;
    const size_t NX = (size_t)BB * SS * DD;   // 4 Mi elements
    const size_t NW = (size_t)DD * DD;        // 1 Mi elements
    size_t off = 0;
    __bf16* xb  = (__bf16*)(ws + off); off += NX * 2;
    __bf16* Wqb = (__bf16*)(ws + off); off += NW * 2;
    __bf16* Wkb = (__bf16*)(ws + off); off += NW * 2;
    __bf16* Wvb = (__bf16*)(ws + off); off += NW * 2;
    __bf16* Wob = (__bf16*)(ws + off); off += NW * 2;
    __bf16* Qb  = (__bf16*)(ws + off); off += NX * 2;
    __bf16* Kb  = (__bf16*)(ws + off); off += NX * 2;
    __bf16* Vt  = (__bf16*)(ws + off); off += NX * 2;
    __bf16* ctx = (__bf16*)(ws + off); off += NX * 2;   // total 48 MiB

    cvt_kernel<<<dim3((unsigned)((NX + 255) / 256)), dim3(256), 0, stream>>>(x,  xb,  (int)NX);
    cvt_kernel<<<dim3((unsigned)((NW + 255) / 256)), dim3(256), 0, stream>>>(Wq, Wqb, (int)NW);
    cvt_kernel<<<dim3((unsigned)((NW + 255) / 256)), dim3(256), 0, stream>>>(Wk, Wkb, (int)NW);
    cvt_kernel<<<dim3((unsigned)((NW + 255) / 256)), dim3(256), 0, stream>>>(Wv, Wvb, (int)NW);
    cvt_kernel<<<dim3((unsigned)((NW + 255) / 256)), dim3(256), 0, stream>>>(Wo, Wob, (int)NW);

    qkv_rope_kernel<<<dim3(BB * SS / 16, 2, 3), dim3(256), 0, stream>>>(
        xb, Wqb, Wkb, Wvb, Qb, Kb, Vt);
    attn_kernel<<<dim3(SS / 16 / 4, HH, BB), dim3(128), 0, stream>>>(Qb, Kb, Vt, ctx);
    oproj_kernel<<<dim3(BB * SS / 16, 2), dim3(256), 0, stream>>>(
        ctx, Wob, (float*)d_out);
}
